// CustomRNN_13726715478735
// MI455X (gfx1250) — matrix operand order repaired and checked by reading, not compiled, not run on hardware
//
#include <hip/hip_runtime.h>

// ---------------------------------------------------------------------------
// 2-layer LSTM forward, persistent-kernel design for MI455X (gfx1250).
//   - f16 WMMA 16x16x32, f32 accumulate (v_wmma_f32_16x16x32_f16)
//   - per-block weight slices resident in 320KB WGP LDS, packed in WMMA B layout
//   - weights staged into LDS by the Tensor Data Mover (tensor_load_to_lds),
//     fenced with s_wait_tensorcnt
//   - c-state held in accumulator-layout VGPRs across all 512 timesteps
//   - double-buffered h broadcast via L2, 2 grid syncs per step
//   - cluster-barrier fast path + global prefetch (CDNA5 features)
// ---------------------------------------------------------------------------

typedef __attribute__((ext_vector_type(16))) _Float16 v16h;
typedef __attribute__((ext_vector_type(8)))  _Float16 v8h;
typedef __attribute__((ext_vector_type(8)))  float    v8f;
typedef __attribute__((ext_vector_type(4)))  unsigned int u32x4;
typedef __attribute__((ext_vector_type(8)))  int      i32x8;
typedef __attribute__((ext_vector_type(4)))  int      i32x4;

#define VOCAB   128
#define HID     512
#define NGATE   2048          // 4*HID
#define BATCH   128
#define TLEN    512
#define NBLK    32            // persistent blocks; block owns 16 h-columns
#define NTHR    256           // 8 waves (wave32) -> 2 waves per SIMD32
#define NCOL    16            // h-columns per block
#define KT0     20            // K=640  (vocab 128 + hid 512) / 32
#define KT1     32            // K=1024 (hid 512 + hid 512) / 32
#define CHUNK0  (4*KT0)       // B-tile chunks per block, layer0
#define CHUNK1  (4*KT1)       // layer1
#define W0_BLK  (CHUNK0*1024) // 80 KB  (each chunk = 32 lanes x 32B)
#define W1_BLK  (CHUNK1*1024) // 128 KB
#define LDSB    (W0_BLK + W1_BLK) // 212992 B  (<= 320KB WGP LDS)

// ---- workspace layout (bytes) ----
#define WS_BAR  ((size_t)0)
#define WS_XH   ((size_t)256)
#define XH_B    ((size_t)BATCH*TLEN*VOCAB*2)         // x in f16, 16 MB
#define WS_W0   (WS_XH + XH_B)
#define WS_W1   (WS_W0 + (size_t)NBLK*W0_BLK)
#define WS_H0   (WS_W1 + (size_t)NBLK*W1_BLK)
#define HB_B    ((size_t)BATCH*HID*2)                // one h buffer, f16
#define WS_H1   (WS_H0 + 2*HB_B)
#define WS_END  (WS_H1 + 2*HB_B)                     // ~24.2 MB total

__device__ __forceinline__ v8f wmma_f16(v16h a, v16h b, v8f c) {
  // (neg_a, A, neg_b, B, c_mod, C, reuse_a, reuse_b)
  return __builtin_amdgcn_wmma_f32_16x16x32_f16(false, a, false, b, (short)0, c,
                                                false, false);
}

__device__ __forceinline__ float sigf(float x) { return 1.f / (1.f + __expf(-x)); }
__device__ __forceinline__ float tanhfast(float x) {
  float e = __expf(-2.f * x);
  return (1.f - e) / (1.f + e);
}

// A-tile fetch, ISA 16-bit A layout: lane<16 holds K {k0..k0+7, k0+16..k0+23}
// for row m = lane; lane>=16 holds K {+8,+24} halves. Caller passes k0 already
// including the per-lane +8 offset.
__device__ __forceinline__ v16h loadA(const _Float16* __restrict__ row, int k0) {
  v8h lo = *(const v8h*)(row + k0);
  v8h hi = *(const v8h*)(row + k0 + 16);
  return __builtin_shufflevector(lo, hi, 0, 1, 2, 3, 4, 5, 6, 7,
                                 8, 9, 10, 11, 12, 13, 14, 15);
}

__device__ __forceinline__ void mm4(const char* __restrict__ lw, int ktn, int kt,
                                    int lane, v16h a,
                                    v8f& a0, v8f& a1, v8f& a2, v8f& a3) {
  a0 = wmma_f16(a, *(const v16h*)(lw + ((size_t)((0 * ktn) + kt) * 32 + lane) * 32), a0);
  a1 = wmma_f16(a, *(const v16h*)(lw + ((size_t)((1 * ktn) + kt) * 32 + lane) * 32), a1);
  a2 = wmma_f16(a, *(const v16h*)(lw + ((size_t)((2 * ktn) + kt) * 32 + lane) * 32), a2);
  a3 = wmma_f16(a, *(const v16h*)(lw + ((size_t)((3 * ktn) + kt) * 32 + lane) * 32), a3);
}

// ---------------------------------------------------------------------------
// Tensor Data Mover: 1-D tile DMA of `bytes` (multiple of 8) from global to
// LDS offset `lds_off`.  D# per ISA ch.8: group0 = {count=1, lds_addr,
// global_addr, type=2}; group1 = {data_size=8B, tensor_dim0 = tile_dim0 = N,
// tensor_dim1 = tile_dim1 = 1, dim0_stride = N}; groups 2/3 unused (2-D mode).
// This toolchain's builtin is the 6-operand form:
//   (u32x4 g0, i32x8 g1, i32x4 g2, i32x4 g3, i32x8 pad, i32 cpol)
// ---------------------------------------------------------------------------
#if __has_builtin(__builtin_amdgcn_tensor_load_to_lds)
#define HAVE_TDM 1
__device__ __forceinline__ void tdm_load_lds(unsigned lds_off,
                                             const void* gptr, unsigned bytes) {
  unsigned long long ga = (unsigned long long)gptr;
  unsigned n8 = bytes >> 3;                       // 8-byte elements
  u32x4 g0;
  g0[0] = 1u;                                     // count=1, user descriptor
  g0[1] = lds_off;                                // lds_addr (bytes)
  g0[2] = (unsigned)(ga & 0xffffffffu);           // global_addr[31:0]
  g0[3] = (unsigned)((ga >> 32) & 0x01ffffffu)    // global_addr[56:32]
        | (2u << 30);                             // type = 2 ("image")
  i32x8 g1;
  g1[0] = (int)(3u << 16);                        // data_size = 3 (8 bytes)
  g1[1] = (int)((n8 & 0xffffu) << 16);            // tensor_dim0[15:0]
  g1[2] = (int)(((n8 >> 16) & 0xffffu)            // tensor_dim0[31:16]
        | (1u << 16));                            // tensor_dim1 = 1 (lo 16)
  g1[3] = (int)((n8 & 0xffffu) << 16);            // tile_dim0 = n8 (fits 16b)
  g1[4] = 1;                                      // tile_dim1 = 1
  g1[5] = (int)n8;                                // tensor_dim0_stride lo32
  g1[6] = 0;                                      // stride hi / dim1_stride lo
  g1[7] = 0;
  i32x4 z4 = {0, 0, 0, 0};                        // groups 2/3: 2-D mode
  i32x8 z8i = {0, 0, 0, 0, 0, 0, 0, 0};
  __builtin_amdgcn_tensor_load_to_lds(g0, g1, z4, z4, z8i, 0);
}
#else
#define HAVE_TDM 0
#endif

// Grid barrier: release stores -> monotonic counter -> acquire. When dispatched
// as a workgroup cluster, also take the cheap cluster split-barrier first.
__device__ __forceinline__ void grid_sync(unsigned* bar, unsigned target) {
  __syncthreads();
#if __has_builtin(__builtin_amdgcn_cluster_id_x) && __has_builtin(__builtin_amdgcn_s_cluster_barrier)
  if (__builtin_amdgcn_cluster_id_x() > 0) __builtin_amdgcn_s_cluster_barrier();
#endif
  if (threadIdx.x == 0) {
    __threadfence();
    __hip_atomic_fetch_add(bar, 1u, __ATOMIC_ACQ_REL, __HIP_MEMORY_SCOPE_AGENT);
    while (__hip_atomic_load(bar, __ATOMIC_ACQUIRE, __HIP_MEMORY_SCOPE_AGENT) < target)
      __builtin_amdgcn_s_sleep(2);
  }
  __syncthreads();
  __threadfence();
}

// ---------------------------------------------------------------------------
// Prologue kernels
// ---------------------------------------------------------------------------

// x (B,T,V) f32 -> f16 (same layout, k-contiguous per (b,t) row)
__global__ void k_pack_x(const float* __restrict__ x, _Float16* __restrict__ xo) {
  size_t i = (size_t)blockIdx.x * blockDim.x + threadIdx.x;
  xo[i] = (_Float16)x[i];
}

// Pack W into per-block, per-gate-tile WMMA B layout:
// chunk (blk, g, kt) -> 1KB: lane l writes 16 f16, n = g*512+blk*16+(l&15),
// k = kt*32 + (l&16?8:0) + (j<8 ? j : j+8). Layer0 K = [vocab | hid], layer1 K = [hid | hid].
__global__ void k_pack_w(const float* __restrict__ Wx0, const float* __restrict__ Wh0,
                         const float* __restrict__ Wx1, const float* __restrict__ Wh1,
                         char* __restrict__ ws) {
  int tid = blockIdx.x * blockDim.x + threadIdx.x;
  int lane = tid & 31;
  int chunk = tid >> 5;                 // 0 .. NBLK*(CHUNK0+CHUNK1)-1
  int blk = chunk / (CHUNK0 + CHUNK1);
  int r = chunk % (CHUNK0 + CHUNK1);
  int nh = blk * NCOL + (lane & 15);
  int koff = (lane & 16) ? 8 : 0;
  v16h v;
  if (r < CHUNK0) {
    int g = r / KT0, kt = r % KT0;
    int n = g * HID + nh;
    int k0 = kt * 32 + koff;
#pragma unroll
    for (int j = 0; j < 16; ++j) {
      int k = k0 + (j < 8 ? j : j + 8);
      float val = (k < VOCAB) ? Wx0[(size_t)k * NGATE + n]
                              : Wh0[(size_t)(k - VOCAB) * NGATE + n];
      v[j] = (_Float16)val;
    }
    *(v16h*)(ws + WS_W0 + (size_t)blk * W0_BLK + ((size_t)r * 32 + lane) * 32) = v;
  } else {
    int r1 = r - CHUNK0;
    int g = r1 / KT1, kt = r1 % KT1;
    int n = g * HID + nh;
    int k0 = kt * 32 + koff;
#pragma unroll
    for (int j = 0; j < 16; ++j) {
      int k = k0 + (j < 8 ? j : j + 8);
      float val = (k < HID) ? Wx1[(size_t)k * NGATE + n]
                            : Wh1[(size_t)(k - HID) * NGATE + n];
      v[j] = (_Float16)val;
    }
    *(v16h*)(ws + WS_W1 + (size_t)blk * W1_BLK + ((size_t)r1 * 32 + lane) * 32) = v;
  }
}

// h0 (L,B,H) f32 -> f16 ping buffers [parity 0]
__global__ void k_init_h(const float* __restrict__ h, char* __restrict__ ws) {
  int i = blockIdx.x * blockDim.x + threadIdx.x;   // 0 .. 2*B*H-1
  int l = i / (BATCH * HID);
  int rem = i - l * (BATCH * HID);
  _Float16* dst = (_Float16*)(ws + (l == 0 ? WS_H0 : WS_H1));
  dst[rem] = (_Float16)h[i];
}

// ---------------------------------------------------------------------------
// Persistent LSTM kernel
// ---------------------------------------------------------------------------
__global__ __launch_bounds__(NTHR, 1)
void k_lstm(const float* __restrict__ c_in, const float* __restrict__ b0p,
            const float* __restrict__ b1p, float* __restrict__ out,
            char* __restrict__ ws) {
  extern __shared__ char lds[];
  const int tid = threadIdx.x;
  const int lane = tid & 31;
  const int wv = tid >> 5;            // 0..7 -> m-tile (rows 16*wv ..)
  const int blk = blockIdx.x;         // 0..31 -> h-cols [16*blk, 16*blk+16)
  const int lhalf = (lane >> 4) & 1;
  const int ln = lane & 15;
  const int ncol = blk * NCOL + ln;   // this lane's h column (C-layout N)
  const int arow = wv * 16 + ln;      // this lane's A row
  const int aoff = lhalf * 8;         // per-lane K offset in A tiles
  const int mbase = wv * 16 + lhalf * 8;  // C-layout: VGPR r -> row mbase+r

  // --- stage this block's packed weight slices into LDS (208 KB) ---
  const char* gw0 = ws + WS_W0 + (size_t)blk * W0_BLK;
  const char* gw1 = ws + WS_W1 + (size_t)blk * W1_BLK;
#if HAVE_TDM
  if (tid < 32) {                     // wave 0 drives the Tensor Data Mover
    tdm_load_lds(0u, gw0, W0_BLK);    // dynamic-LDS base offset is 0 here
    tdm_load_lds((unsigned)W0_BLK, gw1, W1_BLK);
#if __has_builtin(__builtin_amdgcn_s_wait_tensorcnt)
    __builtin_amdgcn_s_wait_tensorcnt(0);
#endif
  }
#else
  for (int i = tid * 16; i < W0_BLK; i += NTHR * 16)
    *(float4*)(lds + i) = *(const float4*)(gw0 + i);
  for (int i = tid * 16; i < W1_BLK; i += NTHR * 16)
    *(float4*)(lds + W0_BLK + i) = *(const float4*)(gw1 + i);
#endif

  // --- biases (per lane: one gate column each of i,f,g,o) ---
  const float bi0 = b0p[0 * HID + ncol], bf0 = b0p[1 * HID + ncol];
  const float bg0 = b0p[2 * HID + ncol], bo0 = b0p[3 * HID + ncol];
  const float bi1 = b1p[0 * HID + ncol], bf1 = b1p[1 * HID + ncol];
  const float bg1 = b1p[2 * HID + ncol], bo1 = b1p[3 * HID + ncol];

  // --- c state in accumulator layout, resident for all 512 steps ---
  v8f c0v, c1v;
#pragma unroll
  for (int r = 0; r < 8; ++r) {
    int m = mbase + r;
    c0v[r] = c_in[(size_t)m * HID + ncol];
    c1v[r] = c_in[(size_t)BATCH * HID + (size_t)m * HID + ncol];
  }

  __syncthreads();                    // LDS weights ready for every wave
  const char* lw0 = lds;
  const char* lw1 = lds + W0_BLK;

  const _Float16* xb = (const _Float16*)(ws + WS_XH);
  _Float16* h0buf[2] = {(_Float16*)(ws + WS_H0), (_Float16*)(ws + WS_H0) + BATCH * HID};
  _Float16* h1buf[2] = {(_Float16*)(ws + WS_H1), (_Float16*)(ws + WS_H1) + BATCH * HID};
  unsigned* bar = (unsigned*)(ws + WS_BAR);
  unsigned tgt = 0;

  const v8f z8 = {0.f, 0.f, 0.f, 0.f, 0.f, 0.f, 0.f, 0.f};
  v8f hf0 = z8, hf1 = z8;

  for (int t = 0; t < TLEN; ++t) {
    const int p = t & 1;
    const _Float16* h0r = h0buf[p];
    _Float16* h0w = h0buf[1 - p];
    const _Float16* h1r = h1buf[p];
    _Float16* h1w = h1buf[1 - p];

    // ================= layer 0: gates = [x_t | h0_{t-1}] @ W0 =================
    v8f acc0 = z8, acc1 = z8, acc2 = z8, acc3 = z8;
    const _Float16* ax = xb + ((size_t)arow * TLEN + t) * VOCAB;
#pragma unroll
    for (int kt = 0; kt < 4; ++kt) {                  // K = vocab segment
      v16h a = loadA(ax, kt * 32 + aoff);
      mm4(lw0, KT0, kt, lane, a, acc0, acc1, acc2, acc3);
    }
    const _Float16* ah0 = h0r + (size_t)arow * HID;
#pragma unroll 4
    for (int kt = 4; kt < KT0; ++kt) {                // K = h0 segment
      v16h a = loadA(ah0, (kt - 4) * 32 + aoff);
      mm4(lw0, KT0, kt, lane, a, acc0, acc1, acc2, acc3);
    }
    if (t + 1 < TLEN)                                 // warm next step's x rows
      __builtin_prefetch(xb + ((size_t)arow * TLEN + t + 1) * VOCAB, 0, 3);

    v8f hv;
#pragma unroll
    for (int r = 0; r < 8; ++r) {
      float iv = sigf(acc0[r] + bi0);
      float fv = sigf(acc1[r] + bf0);
      float gv = tanhfast(acc2[r] + bg0);
      float ov = sigf(acc3[r] + bo0);
      float cc = fv * c0v[r] + iv * gv;
      c0v[r] = cc;
      hv[r] = ov * tanhfast(cc);
    }
#pragma unroll
    for (int r = 0; r < 8; ++r)
      h0w[(size_t)(mbase + r) * HID + ncol] = (_Float16)hv[r];
    if (t == TLEN - 1) hf0 = hv;

    tgt += NBLK;
    grid_sync(bar, tgt);   // h0_t visible everywhere

    // ================= layer 1: gates = [h0_t | h1_{t-1}] @ W1 ================
    acc0 = z8; acc1 = z8; acc2 = z8; acc3 = z8;
    const _Float16* ahA = h0w + (size_t)arow * HID;
#pragma unroll 4
    for (int kt = 0; kt < 16; ++kt) {
      v16h a = loadA(ahA, kt * 32 + aoff);
      mm4(lw1, KT1, kt, lane, a, acc0, acc1, acc2, acc3);
    }
    const _Float16* ahB = h1r + (size_t)arow * HID;
#pragma unroll 4
    for (int kt = 16; kt < KT1; ++kt) {
      v16h a = loadA(ahB, (kt - 16) * 32 + aoff);
      mm4(lw1, KT1, kt, lane, a, acc0, acc1, acc2, acc3);
    }
#pragma unroll
    for (int r = 0; r < 8; ++r) {
      float iv = sigf(acc0[r] + bi1);
      float fv = sigf(acc1[r] + bf1);
      float gv = tanhfast(acc2[r] + bg1);
      float ov = sigf(acc3[r] + bo1);
      float cc = fv * c1v[r] + iv * gv;
      c1v[r] = cc;
      hv[r] = ov * tanhfast(cc);
    }
#pragma unroll
    for (int r = 0; r < 8; ++r) {
      int m = mbase + r;
      h1w[(size_t)m * HID + ncol] = (_Float16)hv[r];
      out[((size_t)m * TLEN + t) * HID + ncol] = hv[r];   // outs (B,T,H)
    }
    if (t == TLEN - 1) hf1 = hv;

    tgt += NBLK;
    grid_sync(bar, tgt);   // h1_t visible everywhere
  }

  // final states: h_f (L,B,H) then c_f (L,B,H), appended after outs
  float* hfp = out + (size_t)BATCH * TLEN * HID;
  float* cfp = hfp + (size_t)2 * BATCH * HID;
#pragma unroll
  for (int r = 0; r < 8; ++r) {
    int m = mbase + r;
    hfp[(size_t)m * HID + ncol] = hf0[r];
    hfp[(size_t)BATCH * HID + (size_t)m * HID + ncol] = hf1[r];
    cfp[(size_t)m * HID + ncol] = c0v[r];
    cfp[(size_t)BATCH * HID + (size_t)m * HID + ncol] = c1v[r];
  }
}

// ---------------------------------------------------------------------------
extern "C" void kernel_launch(void* const* d_in, const int* in_sizes, int n_in,
                              void* d_out, int out_size, void* d_ws, size_t ws_size,
                              hipStream_t stream) {
  (void)in_sizes; (void)n_in; (void)out_size; (void)ws_size;
  const float* x   = (const float*)d_in[0];
  const float* h   = (const float*)d_in[1];
  const float* c   = (const float*)d_in[2];
  const float* Wx0 = (const float*)d_in[3];
  const float* Wh0 = (const float*)d_in[4];
  const float* b0  = (const float*)d_in[5];
  const float* Wx1 = (const float*)d_in[6];
  const float* Wh1 = (const float*)d_in[7];
  const float* b1  = (const float*)d_in[8];
  char* ws = (char*)d_ws;
  float* out = (float*)d_out;

  // barrier counter must start at 0 every call (graph-capturable)
  (void)hipMemsetAsync(ws + WS_BAR, 0, 256, stream);

  k_pack_x<<<(BATCH * TLEN * VOCAB) / 256, 256, 0, stream>>>(x, (_Float16*)(ws + WS_XH));
  k_pack_w<<<(NBLK * (CHUNK0 + CHUNK1) * 32) / 256, 256, 0, stream>>>(Wx0, Wh0, Wx1, Wh1, ws);
  k_init_h<<<(2 * BATCH * HID) / 256, 256, 0, stream>>>(h, ws);

  (void)hipFuncSetAttribute(reinterpret_cast<const void*>(k_lstm),
                            hipFuncAttributeMaxDynamicSharedMemorySize, (int)LDSB);
  k_lstm<<<NBLK, NTHR, LDSB, stream>>>(c, b0, b1, out, ws);
}
